// RealVectorSHT_2731599200380
// MI455X (gfx1250) — matrix operand order repaired and checked by reading, not compiled, not run on hardware
//
#include <hip/hip_runtime.h>
#include <hip/hip_bf16.h>

typedef __attribute__((ext_vector_type(16))) _Float16 v16h;
typedef __attribute__((ext_vector_type(8)))  _Float16 v8h;
typedef __attribute__((ext_vector_type(8)))  float    v8f;

#define NLAT 256
#define NLON 512
#define LMAX 256
#define MMAX 257
#define MPAD 272          // 17 * 16, m padded to a multiple of 16
#define BATCH 16
#define CHAN 8
#define ROWS1 (BATCH*CHAN*2*NLAT)   // 65536 FFT rows
#define BC    (BATCH*CHAN)          // 128 batch rows for the contraction

// ---- workspace layout (bytes) ----
#define OFF_F    0ULL
#define SZ_F     (2ULL*MPAD*NLON*2ULL)                 // f16 DFT matrices [2][m][n]
#define OFF_WH   (OFF_F + SZ_F)
#define SZ_WH    (2ULL*MMAX*LMAX*NLAT*2ULL)            // f16 weights [2][m][l][k]
#define OFF_SPEC (OFF_WH + SZ_WH)                      // f16 spectrum [m][q][bc][lat]

// ---- gfx1250 async memory->LDS path (guarded; falls back to sync staging) ----
#if __has_builtin(__builtin_amdgcn_global_load_async_to_lds_b128) && \
    __has_builtin(__builtin_amdgcn_s_wait_asynccnt)
#define USE_ASYNC_LDS 1
#else
#define USE_ASYNC_LDS 0
#endif

// builtin signature is (int4 AS1* src, int4 AS3* dst, imm offset, imm cpol)
// with the generic-vector (vector_size) int4 type
typedef int v4i_vs __attribute__((vector_size(16)));
typedef __attribute__((address_space(1))) v4i_vs gv4i;
typedef __attribute__((address_space(3))) v4i_vs lv4i;

static __device__ inline void async_b128(const void* g, void* l) {
#if USE_ASYNC_LDS
    __builtin_amdgcn_global_load_async_to_lds_b128((gv4i*)g, (lv4i*)l, 0, 0);
#else
    (void)g; (void)l;
#endif
}

union Frag { v16h v; v8h h[2]; };

static __device__ inline v8f wmma16(v16h a, v16h b, v8f c) {
    return __builtin_amdgcn_wmma_f32_16x16x32_f16(false, a, false, b, (short)0, c,
                                                  false, false);
}

// ---------------- init: f16 DFT matrix, [mat][m][n], n contiguous ----------------
__global__ void k_init_dft(_Float16* __restrict__ F) {
    int idx = blockIdx.x * 256 + threadIdx.x;
    if (idx >= 2 * MPAD * NLON) return;
    int mat = idx / (MPAD * NLON);
    int rem = idx % (MPAD * NLON);
    int m = rem / NLON;
    int n = rem % NLON;
    float val = 0.0f;
    if (m < MMAX) {
        const float scale = 6.28318530717958647692f / (float)NLON; // 2*pi * (1/N)
        float th = 6.28318530717958647692f * (float)((m * n) & (NLON - 1)) / (float)NLON;
        val = (mat == 0) ? scale * cosf(th) : -scale * sinf(th);
    }
    F[idx] = (_Float16)val;
}

// ---------------- init: weights f32 -> f16 (same layout) ----------------
__global__ void k_init_wh(const float* __restrict__ w, _Float16* __restrict__ wh, int n) {
    int i = blockIdx.x * 256 + threadIdx.x;
    if (i < n) wh[i] = (_Float16)w[i];
}

// ---------------- stage 1: rFFT as GEMM on WMMA ----------------
// grid (ROWS1/128, MPAD/16), block 256 (8 waves). Wave w owns a 16-row tile,
// block owns one 16-wide m tile. K loop over NLON in steps of 32.
__global__ void k_dft(const float* __restrict__ x, const _Float16* __restrict__ F,
                      _Float16* __restrict__ spec) {
    __shared__ __align__(16) _Float16 aBuf[128 * 32];    // 8 KB (reused for transpose)
    __shared__ __align__(16) _Float16 bBuf[2 * 16 * 32]; // 2 KB
    const int t = threadIdx.x;
    const int w = t >> 5, lane = t & 31, lsel = lane & 15, hi = lane >> 4;
    const int rowbase = blockIdx.x * 128;
    const int mb = blockIdx.y * 16;

    v8f accR = {}; v8f accI = {};

    for (int kt = 0; kt < 16; ++kt) {
        const int kb = kt * 32;
        __syncthreads();
#if USE_ASYNC_LDS
        // B tiles: pure f16 copy -> async DMA into LDS, overlapped with A convert
        if (t < 128) {
            int c = t, mat = c >> 6, col = (c >> 2) & 15, kc = (c & 3) * 8;
            async_b128(F + ((size_t)mat * MPAD + mb + col) * NLON + kb + kc,
                       &bBuf[(mat * 16 + col) * 32 + kc]);
        }
#endif
        // stage A: 128 rows x 32 k, f32 -> f16 (needs VALU convert)
        #pragma unroll
        for (int i = 0; i < 2; ++i) {
            int c = t + 256 * i;                 // 512 chunks of 8 halves
            int r = c >> 2, kc = (c & 3) * 8;
            const float4* g = (const float4*)(x + (size_t)(rowbase + r) * NLON + kb + kc);
            float4 p = g[0], q = g[1];
            if (kt + 1 < 16) __builtin_prefetch((const void*)(g + 8), 0, 1);
            v8h hh;
            hh[0] = (_Float16)p.x; hh[1] = (_Float16)p.y;
            hh[2] = (_Float16)p.z; hh[3] = (_Float16)p.w;
            hh[4] = (_Float16)q.x; hh[5] = (_Float16)q.y;
            hh[6] = (_Float16)q.z; hh[7] = (_Float16)q.w;
            *(v8h*)&aBuf[r * 32 + kc] = hh;
        }
#if USE_ASYNC_LDS
        __builtin_amdgcn_s_wait_asynccnt(0);
#else
        if (t < 128) {
            int c = t, mat = c >> 6, col = (c >> 2) & 15, kc = (c & 3) * 8;
            const v8h* g = (const v8h*)(F + ((size_t)mat * MPAD + mb + col) * NLON + kb + kc);
            *(v8h*)&bBuf[(mat * 16 + col) * 32 + kc] = g[0];
        }
#endif
        __syncthreads();

        Frag a, br, bi;
        { int r = w * 16 + lsel; int o = hi * 16;
          a.h[0] = *(const v8h*)&aBuf[r * 32 + o];
          a.h[1] = *(const v8h*)&aBuf[r * 32 + o + 8]; }
        { int o = hi * 16;
          br.h[0] = *(const v8h*)&bBuf[(0 * 16 + lsel) * 32 + o];
          br.h[1] = *(const v8h*)&bBuf[(0 * 16 + lsel) * 32 + o + 8];
          bi.h[0] = *(const v8h*)&bBuf[(1 * 16 + lsel) * 32 + o];
          bi.h[1] = *(const v8h*)&bBuf[(1 * 16 + lsel) * 32 + o + 8]; }

        accR = wmma16(a.v, br.v, accR);
        accI = wmma16(a.v, bi.v, accI);
    }

    // ---- transpose D tiles through LDS, then 16B lat-contiguous stores ----
    // block covers: fixed (bc, comp), lat in [rowbase&255, +128), m in [mb, mb+16)
    __syncthreads();                 // all waves done reading aBuf fragments
    {
        _Float16* tb = &aBuf[w * 512];            // per-wave: [reim][16 m][16 lat]
        #pragma unroll
        for (int i = 0; i < 8; ++i) {
            int latl = i + hi * 8;
            tb[lsel * 16 + latl]       = (_Float16)accR[i];
            tb[256 + lsel * 16 + latl] = (_Float16)accI[i];
        }
        __syncthreads();
        const int m_loc = lane >> 1, hs = lane & 1;
        v8h re = *(const v8h*)&tb[m_loc * 16 + hs * 8];
        v8h im = *(const v8h*)&tb[256 + m_loc * 16 + hs * 8];
        const int comp = (rowbase >> 8) & 1;
        const int bc   = rowbase >> 9;
        const int latb = (rowbase & (NLAT - 1)) + w * 16 + hs * 8;
        size_t baseR = (((size_t)(mb + m_loc) * 4 + comp * 2)     * BC + bc) * NLAT + latb;
        size_t baseI = (((size_t)(mb + m_loc) * 4 + comp * 2 + 1) * BC + bc) * NLAT + latb;
        *(v8h*)(spec + baseR) = re;
        *(v8h*)(spec + baseI) = im;
    }
}

// ---------------- stage 2: per-m contraction, 8 WMMA products -> 4 complex outs ----------------
// grid (MMAX, 2): x = m, y = l-half of 128. Block 256 (8 waves). Double-buffered
// async-DMA staging: batch k+1 in flight while computing batch k.
__global__ void k_contract(const _Float16* __restrict__ spec,
                           const _Float16* __restrict__ wh,
                           float* __restrict__ out) {
    __shared__ __align__(16) _Float16 aBuf[2][4 * 128 * 32]; // 2 x 32 KB
    __shared__ __align__(16) _Float16 bBuf[2][2 * 32 * 32];  // 2 x 4 KB
    const int t = threadIdx.x;
    const int w = t >> 5, lane = t & 31, lsel = lane & 15, hi = lane >> 4;
    const int m = blockIdx.x;
    const int lbase = blockIdx.y * 128;
    const _Float16* specm = spec + (size_t)m * 4 * BC * NLAT;

    v8f acc[2][4];

#if USE_ASYNC_LDS
    auto issue = [&](int it) {
        const int kb = (it & 7) * 32;
        const int lb2 = lbase + (it >> 3) * 32;
        const int bb = it & 1;
        #pragma unroll
        for (int i = 0; i < 8; ++i) {
            int c = t + 256 * i;                         // 2048 chunks of 8 halves
            int q = c >> 9, r = (c >> 2) & 127, kc = (c & 3) * 8;
            async_b128(specm + ((size_t)q * BC + r) * NLAT + kb + kc,
                       &aBuf[bb][(q * 128 + r) * 32 + kc]);
        }
        {
            int c = t;
            int mat = c >> 7, col = (c >> 2) & 31, kc = (c & 3) * 8;
            async_b128(wh + (((size_t)mat * MMAX + m) * LMAX + lb2 + col) * NLAT + kb + kc,
                       &bBuf[bb][(mat * 32 + col) * 32 + kc]);
        }
    };
    issue(0);
#endif

    for (int it = 0; it < 32; ++it) {
        const int ltp = it >> 3, kt = it & 7, bb = it & 1;
        const int lb = lbase + ltp * 32;
        if (kt == 0) {
            #pragma unroll
            for (int h2 = 0; h2 < 2; ++h2)
                #pragma unroll
                for (int j = 0; j < 4; ++j) acc[h2][j] = (v8f){};
        }
#if USE_ASYNC_LDS
        if (it + 1 < 32) { issue(it + 1); __builtin_amdgcn_s_wait_asynccnt(9); }
        else             { __builtin_amdgcn_s_wait_asynccnt(0); }
#else
        {
            const int kb = kt * 32;
            #pragma unroll
            for (int i = 0; i < 8; ++i) {
                int c = t + 256 * i;
                int q = c >> 9, r = (c >> 2) & 127, kc = (c & 3) * 8;
                const v8h* g = (const v8h*)(specm + ((size_t)q * BC + r) * NLAT + kb + kc);
                *(v8h*)&aBuf[bb][(q * 128 + r) * 32 + kc] = g[0];
            }
            {
                int c = t;
                int mat = c >> 7, col = (c >> 2) & 31, kc = (c & 3) * 8;
                const v8h* g = (const v8h*)(wh + (((size_t)mat * MMAX + m) * LMAX + lb + col) * NLAT + kb + kc);
                *(v8h*)&bBuf[bb][(mat * 32 + col) * 32 + kc] = g[0];
            }
        }
#endif
        __syncthreads();

        Frag aq[4];
        #pragma unroll
        for (int q = 0; q < 4; ++q) {
            int r = w * 16 + lsel; int o = hi * 16;
            aq[q].h[0] = *(const v8h*)&aBuf[bb][(q * 128 + r) * 32 + o];
            aq[q].h[1] = *(const v8h*)&aBuf[bb][(q * 128 + r) * 32 + o + 8];
        }
        // f16 WMMA NEG bits only support C-negate -> negate A fragments in VALU
        v16h n1 = -aq[1].v;   // -Xti
        v16h n2 = -aq[2].v;   // -Xpr
        v16h n3 = -aq[3].v;   // -Xpi

        #pragma unroll
        for (int h = 0; h < 2; ++h) {
            Frag b0, b1;
            int col = h * 16 + lsel; int o = hi * 16;
            b0.h[0] = *(const v8h*)&bBuf[bb][(0 * 32 + col) * 32 + o];
            b0.h[1] = *(const v8h*)&bBuf[bb][(0 * 32 + col) * 32 + o + 8];
            b1.h[0] = *(const v8h*)&bBuf[bb][(1 * 32 + col) * 32 + o];
            b1.h[1] = *(const v8h*)&bBuf[bb][(1 * 32 + col) * 32 + o + 8];

            acc[h][0] = wmma16(aq[0].v, b0.v, acc[h][0]);  // + Xtr*W0
            acc[h][0] = wmma16(n3,      b1.v, acc[h][0]);  // - Xpi*W1
            acc[h][1] = wmma16(aq[1].v, b0.v, acc[h][1]);  // + Xti*W0
            acc[h][1] = wmma16(aq[2].v, b1.v, acc[h][1]);  // + Xpr*W1
            acc[h][2] = wmma16(n1,      b1.v, acc[h][2]);  // - Xti*W1
            acc[h][2] = wmma16(n2,      b0.v, acc[h][2]);  // - Xpr*W0
            acc[h][3] = wmma16(aq[0].v, b1.v, acc[h][3]);  // + Xtr*W1
            acc[h][3] = wmma16(n3,      b0.v, acc[h][3]);  // - Xpi*W0
        }
        __syncthreads();   // all waves done with buffer bb -> reusable for it+2

        if (kt == 7) {
            // store complex64: re/im adjacent; out stays L2-resident so the 8B
            // granule write-combines across adjacent-m blocks in L2
            #pragma unroll
            for (int h = 0; h < 2; ++h) {
                int l = lb + h * 16 + lsel;
                #pragma unroll
                for (int i = 0; i < 8; ++i) {
                    int rr = w * 16 + i + hi * 8;  // bc row
                    size_t i0 = (((size_t)rr * 2 + 0) * LMAX + l) * MMAX + m;
                    size_t i1 = (((size_t)rr * 2 + 1) * LMAX + l) * MMAX + m;
                    *(float2*)(out + 2 * i0) = make_float2(acc[h][0][i], acc[h][1][i]);
                    *(float2*)(out + 2 * i1) = make_float2(acc[h][2][i], acc[h][3][i]);
                }
            }
        }
    }
}

extern "C" void kernel_launch(void* const* d_in, const int* in_sizes, int n_in,
                              void* d_out, int out_size, void* d_ws, size_t ws_size,
                              hipStream_t stream) {
    (void)in_sizes; (void)n_in; (void)out_size; (void)ws_size;
    const float* x   = (const float*)d_in[0];
    const float* wts = (const float*)d_in[1];
    char* ws = (char*)d_ws;
    _Float16* F    = (_Float16*)(ws + OFF_F);
    _Float16* wh   = (_Float16*)(ws + OFF_WH);
    _Float16* spec = (_Float16*)(ws + OFF_SPEC);
    float* out = (float*)d_out;

    k_init_dft<<<(2 * MPAD * NLON + 255) / 256, 256, 0, stream>>>(F);
    const int nw = 2 * MMAX * LMAX * NLAT;
    k_init_wh<<<(nw + 255) / 256, 256, 0, stream>>>(wts, wh, nw);
    k_dft<<<dim3(ROWS1 / 128, MPAD / 16), 256, 0, stream>>>(x, F, spec);
    k_contract<<<dim3(MMAX, 2), 256, 0, stream>>>(spec, wh, out);
}
